// EGNN_layer_2259152798551
// MI455X (gfx1250) — compile-verified, matrix operand order checked by reading, not run on hardware
//
#include <hip/hip_runtime.h>

// ---------------- CDNA5 wave32 / WMMA types ----------------
typedef __attribute__((ext_vector_type(16))) _Float16 v16h;
typedef __attribute__((ext_vector_type(8)))  _Float16 v8h;
typedef __attribute__((ext_vector_type(8)))  float    v8f;

#define Bn    8
#define Nn    384
#define DIMn  64
#define EDIMn 4
#define EHn   128
#define MDIMn 64
#define NWAVES 8
#define JT    (Nn / 16)   // 24 j-tiles

// LDS working set (~104 KB) -- passed as dynamic shared memory
struct LdsT {
  _Float16 W1 [8][3][32][16];     // We1-effective, B-layout swizzled, f16   (24 KB)
  _Float16 W2 [4][4][32][16];     // We2, B-layout                           (16 KB)
  _Float16 Wc1[4][2][32][16];     // Wc1, B-layout                           ( 8 KB)
  _Float16 h1 [NWAVES][16][EHn];  // per-wave silu(layer1) staging           (32 KB)
  _Float16 ms [NWAVES][16][MDIMn];// per-wave silu(msg) staging              (16 KB)
  alignas(16) float eRow[Nn * EDIMn]; // edges[b,i,:,:] async-staged         ( 6 KB)
  float cbias[EHn];               // be1 + h_i @ We1[:64]
  float featsI[DIMn];
  float msum[MDIMn];              // sum_j m  (atomic)
  float agg[4];                   // sum_j rel*cw (atomic)
  float cwst[NWAVES][16];
  float h1n[EHn];
};

__device__ __forceinline__ float silu_f(float x) { return x / (1.0f + __expf(-x)); }
__device__ __forceinline__ float selu_f(float x) {
  const float a = 1.6732632423543772f, s = 1.0507009873554805f;
  return x > 0.0f ? s * x : s * a * (__expf(x) - 1.0f);
}

// ---- CDNA5 async global->LDS copy (ASYNCcnt-tracked DMA path) ----
__device__ __forceinline__ void async_g2l_b128(const void* gsrc, void* ldst) {
#if defined(__AMDGCN__)
  unsigned lofs = (unsigned)(size_t)ldst;    // generic shared addr low 32 bits = LDS offset
  asm volatile("global_load_async_to_lds_b128 %0, %1, off"
               :: "v"(lofs), "v"(gsrc) : "memory");
#else
  (void)gsrc; (void)ldst;
#endif
}
__device__ __forceinline__ void wait_asynccnt0() {
#if defined(__AMDGCN__)
  asm volatile("s_wait_asynccnt 0x0" ::: "memory");
#endif
}

__device__ __forceinline__ v8f wmma_f16(v16h a, v16h b, v8f c) {
  // (neg_a, A, neg_b, B, c_mod, C, reuse_a, reuse_b)
  return __builtin_amdgcn_wmma_f32_16x16x32_f16(false, a, false, b, (short)0, c, false, false);
}

__global__ void __launch_bounds__(256)
egnn_fused_kernel(const float* __restrict__ feats, const float* __restrict__ coors,
                  const float* __restrict__ vel,   const float* __restrict__ edges,
                  const float* __restrict__ We1,   const float* __restrict__ be1,
                  const float* __restrict__ We2,   const float* __restrict__ be2,
                  const float* __restrict__ Wc1,   const float* __restrict__ bc1,
                  const float* __restrict__ Wc2,   const float* __restrict__ bc2,
                  const float* __restrict__ Wn1,   const float* __restrict__ bn1,
                  const float* __restrict__ Wn2,   const float* __restrict__ bn2,
                  const float* __restrict__ Wv,    const float* __restrict__ bv,
                  float* __restrict__ out) {
  extern __shared__ char smem_raw[];
  LdsT* lds = (LdsT*)smem_raw;

  const int tid  = threadIdx.x;
  const int bi   = blockIdx.x;
  const int bb   = bi / Nn;
  const int i    = bi % Nn;
  const int wave = tid >> 5;        // wave32
  const int l    = tid & 31;
  const int m    = l & 15;          // row within 16-tile
  const int kb   = (l >> 4) * 8;    // K sub-base per A/B layout
  const int mrow = (l >> 4) * 8;    // C-layout row base

  // ---------------- phase A: async-stage edge row, swizzle weights ----------------
  // edges[b,i,:,0:4] = 6 KB contiguous, 16B aligned -> one b128 async copy / 16B,
  // overlapped with the f16 weight-convert/swizzle below.
  {
    const float* esrc = edges + (size_t)(bb * Nn + i) * Nn * EDIMn;
    for (int idx = tid; idx < (Nn * EDIMn) / 4; idx += 256)
      async_g2l_b128(esrc + idx * 4, &lds->eRow[idx * 4]);
  }

  if (tid < DIMn) lds->featsI[tid] = feats[(size_t)(bb * Nn + i) * DIMn + tid];
  if (tid < MDIMn) lds->msum[tid] = 0.0f;
  if (tid < 4)     lds->agg[tid]  = 0.0f;

  // We1-effective (varying part): row K<64 -> h_j rows (We1 rows 64..127),
  // K==64 -> dist2 row (128), K 65..68 -> edge rows (129..132), else 0.
  for (int idx = tid; idx < 8 * 3 * 32 * 16; idx += 256) {
    int h = idx & 15, ll = (idx >> 4) & 31, t2 = idx >> 9;
    int ks = t2 % 3, nt = t2 / 3;
    int K = 32 * ks + ((ll >> 4) * 8) + (h & 7) + ((h >> 3) * 16);
    int n = nt * 16 + (ll & 15);
    float v = (K < 69) ? We1[(size_t)(64 + K) * EHn + n] : 0.0f;
    lds->W1[nt][ks][ll][h] = (_Float16)v;
  }
  for (int idx = tid; idx < 4 * 4 * 32 * 16; idx += 256) {
    int h = idx & 15, ll = (idx >> 4) & 31, t2 = idx >> 9;
    int ks = t2 & 3, nt = t2 >> 2;
    int K = 32 * ks + ((ll >> 4) * 8) + (h & 7) + ((h >> 3) * 16);
    int n = nt * 16 + (ll & 15);
    lds->W2[nt][ks][ll][h] = (_Float16)We2[(size_t)K * MDIMn + n];
  }
  for (int idx = tid; idx < 4 * 2 * 32 * 16; idx += 256) {
    int h = idx & 15, ll = (idx >> 4) & 31, t2 = idx >> 9;
    int ks = t2 & 1, nt = t2 >> 1;
    int K = 32 * ks + ((ll >> 4) * 8) + (h & 7) + ((h >> 3) * 16);
    int n = nt * 16 + (ll & 15);
    lds->Wc1[nt][ks][ll][h] = (_Float16)Wc1[(size_t)K * MDIMn + n];
  }
  wait_asynccnt0();                 // edge-row DMA complete (this wave's issues)
  __syncthreads();                  // ... and visible workgroup-wide

  // ---------------- phase B: per-i bias vector (folds h_i @ We1_top) ----------------
  if (tid < EHn) {
    float acc = be1[tid];
    for (int k = 0; k < DIMn; ++k) acc += lds->featsI[k] * We1[(size_t)k * EHn + tid];
    lds->cbias[tid] = acc;
  }
  __syncthreads();

  // per-lane constants
  float be2l[4], bc1l[4], wc2l[4], cbl[8];
  #pragma unroll
  for (int nt = 0; nt < 4; ++nt) {
    be2l[nt] = be2[nt * 16 + (l & 15)];
    bc1l[nt] = bc1[nt * 16 + (l & 15)];
    wc2l[nt] = Wc2[nt * 16 + (l & 15)];
  }
  #pragma unroll
  for (int nt = 0; nt < 8; ++nt) cbl[nt] = lds->cbias[nt * 16 + (l & 15)];
  const float bc2v = bc2[0];
  const float ciX = coors[(size_t)(bb * Nn + i) * 3 + 0];
  const float ciY = coors[(size_t)(bb * Nn + i) * 3 + 1];
  const float ciZ = coors[(size_t)(bb * Nn + i) * 3 + 2];

  // ---------------- main loop: 16-j tiles per wave ----------------
  for (int jt = wave; jt < JT; jt += NWAVES) {
    const int j0 = jt * 16;
    const int j  = j0 + m;
    if (jt + NWAVES < JT)
      __builtin_prefetch(feats + (size_t)(bb * Nn + (j0 + NWAVES * 16)) * DIMn, 0, 1);

    // ---- A tiles (A-layout: lane=row m; halves0-7 -> K kb..kb+7, 8-15 -> K kb+16..kb+23) ----
    union { v16h v; _Float16 h[16]; } a0, a1, a2;
    const float* frow = feats + (size_t)(bb * Nn + j) * DIMn;
    #pragma unroll
    for (int q = 0; q < 8; ++q) { a0.h[q] = (_Float16)frow[kb + q];      a0.h[8 + q] = (_Float16)frow[kb + 16 + q]; }
    #pragma unroll
    for (int q = 0; q < 8; ++q) { a1.h[q] = (_Float16)frow[32 + kb + q]; a1.h[8 + q] = (_Float16)frow[32 + kb + 16 + q]; }
    float relx = 0.0f, rely = 0.0f, relz = 0.0f;
    #pragma unroll
    for (int q = 0; q < 16; ++q) a2.h[q] = (_Float16)0.0f;
    if (l < 16) {               // only K rows 64..71 are non-zero (dist2 + 4 edge feats)
      const float* cj = coors + (size_t)(bb * Nn + j) * 3;
      relx = ciX - cj[0]; rely = ciY - cj[1]; relz = ciZ - cj[2];
      float d2 = relx * relx + rely * rely + relz * relz;
      const float* er = &lds->eRow[j * EDIMn];     // async-staged in LDS
      a2.h[0] = (_Float16)d2;
      a2.h[1] = (_Float16)er[0]; a2.h[2] = (_Float16)er[1];
      a2.h[3] = (_Float16)er[2]; a2.h[4] = (_Float16)er[3];
    }

    // ---- edge layer 1: [16x96] @ [96x128] ----
    v8f cc[8] = {};
    #pragma unroll
    for (int nt = 0; nt < 8; ++nt) {
      cc[nt] = wmma_f16(a0.v, *(const v16h*)(&lds->W1[nt][0][l][0]), cc[nt]);
      cc[nt] = wmma_f16(a1.v, *(const v16h*)(&lds->W1[nt][1][l][0]), cc[nt]);
      cc[nt] = wmma_f16(a2.v, *(const v16h*)(&lds->W1[nt][2][l][0]), cc[nt]);
    }
    #pragma unroll
    for (int nt = 0; nt < 8; ++nt) {
      #pragma unroll
      for (int r = 0; r < 8; ++r)
        lds->h1[wave][mrow + r][nt * 16 + (l & 15)] = (_Float16)silu_f(cc[nt][r] + cbl[nt]);
    }

    // ---- edge layer 2: [16x128] @ [128x64] ----
    v8f mm[4] = {};
    #pragma unroll
    for (int ks = 0; ks < 4; ++ks) {
      v8h lo = *(const v8h*)(&lds->h1[wave][m][32 * ks + kb]);
      v8h hi = *(const v8h*)(&lds->h1[wave][m][32 * ks + kb + 16]);
      v16h av = __builtin_shufflevector(lo, hi, 0,1,2,3,4,5,6,7,8,9,10,11,12,13,14,15);
      #pragma unroll
      for (int nt = 0; nt < 4; ++nt)
        mm[nt] = wmma_f16(av, *(const v16h*)(&lds->W2[nt][ks][l][0]), mm[nt]);
    }
    // silu -> messages; accumulate sum_j m; stage for coor-weight GEMM
    #pragma unroll
    for (int nt = 0; nt < 4; ++nt) {
      float psum = 0.0f;
      #pragma unroll
      for (int r = 0; r < 8; ++r) {
        float sm = silu_f(mm[nt][r] + be2l[nt]);
        lds->ms[wave][mrow + r][nt * 16 + (l & 15)] = (_Float16)sm;
        psum += sm;
      }
      atomicAdd(&lds->msum[nt * 16 + (l & 15)], psum);
    }

    // ---- coor-weight hidden: [16x64] @ [64x64] ----
    v8f hc[4] = {};
    #pragma unroll
    for (int ks = 0; ks < 2; ++ks) {
      v8h lo = *(const v8h*)(&lds->ms[wave][m][32 * ks + kb]);
      v8h hi = *(const v8h*)(&lds->ms[wave][m][32 * ks + kb + 16]);
      v16h av = __builtin_shufflevector(lo, hi, 0,1,2,3,4,5,6,7,8,9,10,11,12,13,14,15);
      #pragma unroll
      for (int nt = 0; nt < 4; ++nt)
        hc[nt] = wmma_f16(av, *(const v16h*)(&lds->Wc1[nt][ks][l][0]), hc[nt]);
    }
    // cw = silu(hc) @ Wc2 + bc2 : 64->1 dot via cross-lane reduction (width-16 groups)
    #pragma unroll
    for (int r = 0; r < 8; ++r) {
      float t = 0.0f;
      #pragma unroll
      for (int nt = 0; nt < 4; ++nt) t += silu_f(hc[nt][r] + bc1l[nt]) * wc2l[nt];
      t += __shfl_xor(t, 1, 32);
      t += __shfl_xor(t, 2, 32);
      t += __shfl_xor(t, 4, 32);
      t += __shfl_xor(t, 8, 32);
      if ((l & 15) == 0) lds->cwst[wave][mrow + r] = t + bc2v;
    }

    // ---- agg += rel * cw (lanes >= 16 carry rel == 0) ----
    float cwm = lds->cwst[wave][m];
    float px = relx * cwm, py = rely * cwm, pz = relz * cwm;
    #pragma unroll
    for (int d = 1; d < 32; d <<= 1) {
      px += __shfl_xor(px, d, 32);
      py += __shfl_xor(py, d, 32);
      pz += __shfl_xor(pz, d, 32);
    }
    if (l == 0) {
      atomicAdd(&lds->agg[0], px);
      atomicAdd(&lds->agg[1], py);
      atomicAdd(&lds->agg[2], pz);
    }
  }
  __syncthreads();

  // ---------------- epilogue: node update (matvec-sized; VALU) ----------------
  if (tid < EHn) {
    float acc = bn1[tid];
    for (int k = 0; k < DIMn; ++k) acc += lds->featsI[k] * Wn1[(size_t)k * (2 * DIMn) + tid];
    for (int k = 0; k < MDIMn; ++k) acc += lds->msum[k] * Wn1[(size_t)(DIMn + k) * (2 * DIMn) + tid];
    lds->h1n[tid] = silu_f(acc);
  }
  __syncthreads();
  const size_t OFF_C = (size_t)Bn * Nn * DIMn;
  const size_t OFF_V = OFF_C + (size_t)Bn * Nn * 3;
  if (tid < DIMn) {
    float acc = bn2[tid];
    for (int n = 0; n < 2 * DIMn; ++n) acc += lds->h1n[n] * Wn2[(size_t)n * DIMn + tid];
    out[(size_t)(bb * Nn + i) * DIMn + tid] = lds->featsI[tid] + acc;
  }
  if (tid < 3) {
    float g = bv[0];
    for (int k = 0; k < DIMn; ++k) g += lds->featsI[k] * Wv[k];
    float aggm = lds->agg[tid] * (1.0f / (float)Nn);
    size_t base = (size_t)(bb * Nn + i) * 3 + tid;
    float vn = g * vel[base] + aggm;
    float cn = coors[base] + vn;
    out[OFF_C + base] = selu_f(cn);
    out[OFF_V + base] = selu_f(vn);
  }
}

extern "C" void kernel_launch(void* const* d_in, const int* in_sizes, int n_in,
                              void* d_out, int out_size, void* d_ws, size_t ws_size,
                              hipStream_t stream) {
  (void)in_sizes; (void)n_in; (void)out_size; (void)d_ws; (void)ws_size;
  const float* feats = (const float*)d_in[0];
  const float* coors = (const float*)d_in[1];
  const float* vel   = (const float*)d_in[2];
  const float* edges = (const float*)d_in[3];
  const float* We1 = (const float*)d_in[4];  const float* be1 = (const float*)d_in[5];
  const float* We2 = (const float*)d_in[6];  const float* be2 = (const float*)d_in[7];
  const float* Wc1 = (const float*)d_in[8];  const float* bc1 = (const float*)d_in[9];
  const float* Wc2 = (const float*)d_in[10]; const float* bc2 = (const float*)d_in[11];
  const float* Wn1 = (const float*)d_in[12]; const float* bn1 = (const float*)d_in[13];
  const float* Wn2 = (const float*)d_in[14]; const float* bn2 = (const float*)d_in[15];
  const float* Wv  = (const float*)d_in[16]; const float* bv  = (const float*)d_in[17];

  dim3 grid(Bn * Nn);      // one workgroup per (b, i)
  dim3 block(256);         // 8 wave32
  size_t shmem = sizeof(LdsT);
  hipLaunchKernelGGL(egnn_fused_kernel, grid, block, shmem, stream,
                     feats, coors, vel, edges, We1, be1, We2, be2,
                     Wc1, bc1, Wc2, bc2, Wn1, bn1, Wn2, bn2, Wv, bv,
                     (float*)d_out);
}
